// GCNLayer_35467839930437
// MI455X (gfx1250) — compile-verified, matrix operand order checked by reading, not run on hardware
//
#include <hip/hip_runtime.h>
#include <hip/hip_bf16.h>

// ---------------- types for WMMA operands ----------------
typedef __attribute__((ext_vector_type(16))) __bf16 bf16x16;
typedef __attribute__((ext_vector_type(8)))  __bf16 bf16x8;
typedef __attribute__((ext_vector_type(8)))  float  f32x8;
typedef __attribute__((ext_vector_type(4)))  float  f32x4;

#define BATCH 8
#define NDIM  2048   // graph nodes (rows of a / x)
#define FDIM  512    // input features
#define DDIM  512    // output features
#define ROWS  64     // output rows per workgroup
#define KC    32     // K-chunk per WMMA step (bf16 K=32)
#define NCHUNK1 (NDIM / KC)   // 64
#define NCHUNK2 (FDIM / KC)   // 16

union FragAB {
    bf16x16 v;
    bf16x8  h[2];
};
union PackBF2 {
    __bf16   h[2];
    unsigned u;
};

// stage a [KC x 512] f32 chunk, transposed + converted, into Bbuf[f][k].
// Each thread owns a K-pair (2 rows) x 32 features -> packed b32 LDS stores.
__device__ __forceinline__ void stage_b_chunk(const float* __restrict__ src,
                                              __bf16 (*__restrict__ Bbuf)[KC],
                                              int tid)
{
    const int kp = tid >> 4;           // 0..15 : pair of K rows
    const int f0 = (tid & 15) * 32;    // 32 features per thread
    const float* p0 = src + (size_t)(2 * kp) * FDIM + f0;
    const float* p1 = p0 + FDIM;
    #pragma unroll
    for (int i = 0; i < 32; i += 4) {
        f32x4 u = *(const f32x4*)(p0 + i);
        f32x4 w = *(const f32x4*)(p1 + i);
        PackBF2 q0, q1, q2, q3;
        q0.h[0] = (__bf16)u.x; q0.h[1] = (__bf16)w.x;
        q1.h[0] = (__bf16)u.y; q1.h[1] = (__bf16)w.y;
        q2.h[0] = (__bf16)u.z; q2.h[1] = (__bf16)w.z;
        q3.h[0] = (__bf16)u.w; q3.h[1] = (__bf16)w.w;
        *(unsigned*)&Bbuf[f0 + i + 0][2 * kp] = q0.u;
        *(unsigned*)&Bbuf[f0 + i + 1][2 * kp] = q1.u;
        *(unsigned*)&Bbuf[f0 + i + 2][2 * kp] = q2.u;
        *(unsigned*)&Bbuf[f0 + i + 3][2 * kp] = q3.u;
    }
}

// stage a [ROWS x KC] f32 chunk of `a` into Abuf[r][k] (one b128 LDS store).
__device__ __forceinline__ void stage_a_chunk(const float* __restrict__ src,
                                              __bf16 (*__restrict__ Abuf)[KC],
                                              int tid)
{
    const int r  = tid >> 2;
    const int k0 = (tid & 3) * 8;
    const float* ap = src + (size_t)r * NDIM + k0;
    f32x4 v0 = *(const f32x4*)(ap);
    f32x4 v1 = *(const f32x4*)(ap + 4);
    bf16x8 pk;
    pk[0] = (__bf16)v0.x; pk[1] = (__bf16)v0.y; pk[2] = (__bf16)v0.z; pk[3] = (__bf16)v0.w;
    pk[4] = (__bf16)v1.x; pk[5] = (__bf16)v1.y; pk[6] = (__bf16)v1.z; pk[7] = (__bf16)v1.w;
    *(bf16x8*)&Abuf[r][k0] = pk;
}

__global__ __launch_bounds__(256)
void gcn_fused_kernel(const float* __restrict__ x,
                      const float* __restrict__ a,
                      const float* __restrict__ kern,
                      float* __restrict__ out)
{
    // LDS budget: 64KB T + 2x32KB B staging + 2x4KB A staging + 1KB mask = 137KB
    __shared__ alignas(16) __bf16 T_lds[ROWS][FDIM];
    __shared__ alignas(16) __bf16 Bst[2][FDIM][KC];
    __shared__ alignas(16) __bf16 Ast[2][ROWS][KC];
    __shared__ alignas(16) float  mask_part[ROWS][4];

    const int tid  = threadIdx.x;
    const int lane = tid & 31;
    const int wave = tid >> 5;
    const int b    = blockIdx.y;
    const int rowBase = blockIdx.x * ROWS;

    // wave tiling: 4 row strips x 2 column halves; 16 col-tiles (256 cols) per wave
    const int rstrip = (wave & 3) * 16;
    const int chalf  = (wave >> 2);
    const int half   = lane >> 4;   // half-wave (K-group select)
    const int lrow   = lane & 15;   // M (A/C) or N (B/C) index within tile
    const int arow   = rstrip + lrow;
    const int cbase  = chalf * 256 + lrow;

    // ---- row-mask partials: all 256 threads, 4 partials per row ----
    {
        const int r = tid >> 2, qtr = tid & 3;
        const float* xr = x + ((size_t)b * NDIM + rowBase + r) * FDIM + qtr * 128;
        float any = 0.0f;
        #pragma unroll 4
        for (int i = 0; i < 128; i += 4) {
            f32x4 v4 = *(const f32x4*)(xr + i);
            any += fabsf(v4.x) + fabsf(v4.y) + fabsf(v4.z) + fabsf(v4.w);
        }
        mask_part[r][qtr] = any;
    }

    f32x8 acc[16];
    #pragma unroll
    for (int j = 0; j < 16; ++j) acc[j] = f32x8{0, 0, 0, 0, 0, 0, 0, 0};

    const float* aBase = a + ((size_t)b * NDIM + rowBase) * (size_t)NDIM;
    const float* xBase = x + (size_t)b * NDIM * FDIM;

    // ============ Phase 1: T = a_tile @ x[b]   (K = NDIM), double-buffered ============
    stage_a_chunk(aBase, Ast[0], tid);
    stage_b_chunk(xBase, Bst[0], tid);
    __syncthreads();

    for (int c = 0; c < NCHUNK1; ++c) {
        const int cur = c & 1;
        if (c + 1 < NCHUNK1) {   // prefetch next chunk into the other buffer
            stage_a_chunk(aBase + (size_t)(c + 1) * KC, Ast[cur ^ 1], tid);
            stage_b_chunk(xBase + (size_t)(c + 1) * KC * FDIM, Bst[cur ^ 1], tid);
        }

        FragAB af;
        af.h[0] = *(const bf16x8*)&Ast[cur][arow][half * 8];
        af.h[1] = *(const bf16x8*)&Ast[cur][arow][16 + half * 8];
        #pragma unroll
        for (int j = 0; j < 16; ++j) {
            const __bf16* bp = &Bst[cur][cbase + j * 16][0];
            FragAB bfr;
            bfr.h[0] = *(const bf16x8*)(bp + half * 8);
            bfr.h[1] = *(const bf16x8*)(bp + 16 + half * 8);
            acc[j] = __builtin_amdgcn_wmma_f32_16x16x32_bf16(
                false, af.v, false, bfr.v, (short)0, acc[j], false, false);
        }
        __syncthreads();
    }

    // spill T tile to LDS as bf16 (C/D layout: lane half selects M+8)
    #pragma unroll
    for (int j = 0; j < 16; ++j) {
        const int f = cbase + j * 16;
        #pragma unroll
        for (int v = 0; v < 8; ++v) {
            T_lds[rstrip + v + 8 * half][f] = (__bf16)acc[j][v];
        }
    }
    #pragma unroll
    for (int j = 0; j < 16; ++j) acc[j] = f32x8{0, 0, 0, 0, 0, 0, 0, 0};

    // ============ Phase 2: OUT = T @ kernel   (K = FDIM), double-buffered ============
    stage_b_chunk(kern, Bst[0], tid);
    __syncthreads();   // fences T_lds writes + Bst[0] staging

    for (int c = 0; c < NCHUNK2; ++c) {
        const int cur = c & 1;
        if (c + 1 < NCHUNK2) {
            stage_b_chunk(kern + (size_t)(c + 1) * KC * DDIM, Bst[cur ^ 1], tid);
        }

        FragAB af;
        const __bf16* tp = &T_lds[arow][c * KC];
        af.h[0] = *(const bf16x8*)(tp + half * 8);
        af.h[1] = *(const bf16x8*)(tp + 16 + half * 8);
        #pragma unroll
        for (int j = 0; j < 16; ++j) {
            const __bf16* bp = &Bst[cur][cbase + j * 16][0];
            FragAB bfr;
            bfr.h[0] = *(const bf16x8*)(bp + half * 8);
            bfr.h[1] = *(const bf16x8*)(bp + 16 + half * 8);
            acc[j] = __builtin_amdgcn_wmma_f32_16x16x32_bf16(
                false, af.v, false, bfr.v, (short)0, acc[j], false, false);
        }
        __syncthreads();
    }

    // ---- epilogue: ReLU * rowmask, f32 store ----
    float mrow[8];
    #pragma unroll
    for (int v = 0; v < 8; ++v) {
        f32x4 mp = *(const f32x4*)&mask_part[rstrip + v + 8 * half][0];
        mrow[v] = (mp.x + mp.y + mp.z + mp.w) > 0.0f ? 1.0f : 0.0f;
    }

    #pragma unroll
    for (int j = 0; j < 16; ++j) {
        const int d = cbase + j * 16;
        #pragma unroll
        for (int v = 0; v < 8; ++v) {
            const int m = rstrip + v + 8 * half;
            const float val = fmaxf(acc[j][v], 0.0f) * mrow[v];
            out[((size_t)b * NDIM + rowBase + m) * DDIM + d] = val;
        }
    }
}

extern "C" void kernel_launch(void* const* d_in, const int* in_sizes, int n_in,
                              void* d_out, int out_size, void* d_ws, size_t ws_size,
                              hipStream_t stream) {
    // setup_inputs order: x [B,N,F] f32, a [B,N,N] f32, kernel [F,D] f32
    const float* x    = (const float*)d_in[0];
    const float* a    = (const float*)d_in[1];
    const float* kern = (const float*)d_in[2];
    float* out        = (float*)d_out;

    dim3 grid(NDIM / ROWS, BATCH);  // 32 x 8 = 256 workgroups
    dim3 block(256);                // 8 wave32s
    gcn_fused_kernel<<<grid, block, 0, stream>>>(x, a, kern, out);
}
